// StreamingSafetyHead_52965536694907
// MI455X (gfx1250) — matrix operand order repaired and checked by reading, not compiled, not run on hardware
//
#include <hip/hip_runtime.h>
#include <hip/hip_bf16.h>
#include <math.h>

// ---------------------------------------------------------------------------
// StreamingSafetyHead for MI455X (gfx1250): bf16 WMMA + async-LDS + TDM.
// ---------------------------------------------------------------------------

typedef __attribute__((ext_vector_type(16))) __bf16        v16bf;
typedef __attribute__((ext_vector_type(8)))  float         v8f;
typedef __attribute__((ext_vector_type(4)))  unsigned int  u32x4;
typedef __attribute__((ext_vector_type(8)))  int           i32x8;
typedef __attribute__((ext_vector_type(4)))  int           i32x4;
typedef __attribute__((ext_vector_type(4)))  float         f32x4;
typedef __attribute__((ext_vector_type(4)))  unsigned short u16x4;

union FragBF { u32x4 q[2]; v16bf v; };

__device__ inline unsigned short f2bf(float f) {
  unsigned u = __builtin_bit_cast(unsigned, f);
  unsigned r = u + 0x7FFFu + ((u >> 16) & 1u);   // RNE
  return (unsigned short)(r >> 16);
}
__device__ inline float bf2f(unsigned short h) {
  return __builtin_bit_cast(float, (unsigned)h << 16);
}
__device__ inline v8f wmma_bf16(const FragBF& a, const FragBF& b, v8f c) {
  return __builtin_amdgcn_wmma_f32_16x16x32_bf16(false, a.v, false, b.v,
                                                 (short)0, c, false, false);
}

constexpr int B_  = 4;
constexpr int S_  = 2048;
constexpr int DIN = 4096;
constexpr int DP  = 512;
constexpr int DM  = 512;
constexpr int AST = 512;
constexpr int RUN = S_ - AST;   // 1536
constexpr int NSLICE = 8;       // workgroups cooperating on the scan

enum { M_BF16ROW = 0, M_VT = 1, M_F32SCALE = 2, M_GPROJ = 3 };

// ---------------------------------------------------------------------------
// small helper kernels
// ---------------------------------------------------------------------------
__global__ void init_kernel(float* out, const float* bmem,
                            unsigned short* hb, unsigned short* rhb,
                            unsigned* ctrs) {
  int i = blockIdx.x * 256 + threadIdx.x;
  if (i < B_ * RUN) out[i] = bmem[0];
  if (i < 16 * DM) { hb[i] = 0; rhb[i] = 0; }
  if (i < 2) ctrs[i] = 0;
}

__global__ void conv_bf16_kernel(const float* __restrict__ in,
                                 unsigned short* __restrict__ out, long n4) {
  long i = (long)blockIdx.x * blockDim.x + threadIdx.x;
  if (i < n4) {
    f32x4 f = *(const f32x4*)(in + i * 4);
    u16x4 o;
    o.x = f2bf(f.x); o.y = f2bf(f.y); o.z = f2bf(f.z); o.w = f2bf(f.w);
    *(u16x4*)(out + i * 4) = o;
  }
}

// out[c][r] = bf16(in[r][c]);  R, C multiples of 32
__global__ void transpose_bf16_kernel(const float* __restrict__ in,
                                      unsigned short* __restrict__ out,
                                      int R, int C) {
  __shared__ unsigned short tile[32][33];
  const int tx = threadIdx.x & 31;
  const int ty = threadIdx.x >> 5;      // 0..7
  const int rb = blockIdx.x * 32;
  const int cb = blockIdx.y * 32;
  for (int i = 0; i < 4; ++i) {
    int r = rb + ty + i * 8;
    tile[ty + i * 8][tx] = f2bf(in[(long)r * C + cb + tx]);
  }
  __syncthreads();
  for (int i = 0; i < 4; ++i) {
    int c = cb + ty + i * 8;
    out[(long)c * R + rb + tx] = tile[tx][ty + i * 8];
  }
}

__global__ void bias_stage_kernel(const float* bq, const float* bk, const float* bv,
                                  const float* bz, const float* br, const float* bh,
                                  float* bias3, float* biasg) {
  int i = blockIdx.x * 256 + threadIdx.x;
  if (i < DP) {
    bias3[i] = bq[i]; bias3[DP + i] = bk[i]; bias3[2 * DP + i] = bv[i];
    biasg[i] = bz[i]; biasg[DP + i] = br[i]; biasg[2 * DP + i] = bh[i];
  }
}

// ---------------------------------------------------------------------------
// generic bf16 WMMA GEMM: C(M x N) = A(row-major [m][k]) * B([n][k]) (+bias)
// block = 256 threads = 8 waves, WG tile 256(M) x 128(N), K chunked by 32.
// B chunks are double-buffered in LDS via global_load_async_to_lds_b128.
// ---------------------------------------------------------------------------
__device__ inline const unsigned short* gemm_arow(const unsigned short* A,
                                                  long lda, int r, int mode) {
  if (mode == M_GPROJ) {             // row r -> feat[b][AST + t][:]
    int b = r & 3, t = r >> 2;
    return A + ((long)(b * S_ + AST + t)) * DP;
  }
  return A + (long)r * lda;
}

__launch_bounds__(256)
__global__ void gemm_bf16_kernel(const unsigned short* __restrict__ A, long lda, long Az,
                                 const unsigned short* __restrict__ Bm, long ldb, long Bz,
                                 const float* __restrict__ bias, long biasz,
                                 void* __restrict__ out, long outz, long ldo,
                                 int K, int mode, float scale) {
  __shared__ unsigned short Bs[2][128 * 32];
  const int tid  = threadIdx.x;
  const int w    = tid >> 5;
  const int lane = tid & 31;
  const int g    = lane >> 4;
  const int ln   = lane & 15;
  const int z    = blockIdx.z;
  A  += (long)z * Az;
  Bm += (long)z * Bz;
  if (bias) bias += (long)z * biasz;
  const int mbase = blockIdx.x * 256 + w * 32;
  const int nbase = blockIdx.y * 128;

  v8f acc[2][8];
  for (int i = 0; i < 2; ++i)
    for (int j = 0; j < 8; ++j) acc[i][j] = {};

  const unsigned short* arp0 = gemm_arow(A, lda, mbase + ln, mode);
  const unsigned short* arp1 = gemm_arow(A, lda, mbase + 16 + ln, mode);

  // this thread's two async-copy slots (i and i+256 of 512 16B quads per chunk)
  const unsigned ldsb0 = (unsigned)(size_t)(void*)&Bs[0][0];
  const unsigned ldsb1 = (unsigned)(size_t)(void*)&Bs[1][0];
  const int i0 = tid, i1 = tid + 256;
  const long boff0 = (long)(nbase + (i0 >> 2)) * ldb + (i0 & 3) * 8;
  const long boff1 = (long)(nbase + (i1 >> 2)) * ldb + (i1 & 3) * 8;

  auto stage = [&](int kc, unsigned ldsbase) {
    unsigned long long ga0 = (unsigned long long)(const void*)(Bm + boff0 + kc);
    unsigned long long ga1 = (unsigned long long)(const void*)(Bm + boff1 + kc);
    unsigned la0 = ldsbase + (unsigned)i0 * 16u;
    unsigned la1 = ldsbase + (unsigned)i1 * 16u;
    asm volatile("global_load_async_to_lds_b128 %0, %1, off"
                 :: "v"(la0), "v"(ga0) : "memory");
    asm volatile("global_load_async_to_lds_b128 %0, %1, off"
                 :: "v"(la1), "v"(ga1) : "memory");
  };

  stage(0, ldsb0);
  int p = 0;
  for (int kc = 0; kc < K; kc += 32) {
    if (kc + 32 < K) {
      stage(kc + 32, p ? ldsb0 : ldsb1);             // prefetch next chunk
      asm volatile("s_wait_asynccnt 0x2" ::: "memory");  // current chunk done
    } else {
      asm volatile("s_wait_asynccnt 0x0" ::: "memory");
    }
    __syncthreads();

    __builtin_prefetch(arp0 + kc + 32, 0, 0);
    __builtin_prefetch(arp1 + kc + 32, 0, 0);

    FragBF a0, a1;
    a0.q[0] = *(const u32x4*)(arp0 + kc + 8 * g);
    a0.q[1] = *(const u32x4*)(arp0 + kc + 16 + 8 * g);
    a1.q[0] = *(const u32x4*)(arp1 + kc + 8 * g);
    a1.q[1] = *(const u32x4*)(arp1 + kc + 16 + 8 * g);

    const unsigned short* Bp = &Bs[p][0];
    for (int half = 0; half < 2; ++half) {
      FragBF bf4[4];
      for (int j = 0; j < 4; ++j) {
        const unsigned short* bp = Bp + ((half * 4 + j) * 16 + ln) * 32 + 8 * g;
        bf4[j].q[0] = *(const u32x4*)bp;
        bf4[j].q[1] = *(const u32x4*)(bp + 16);
      }
      for (int j = 0; j < 4; ++j) {
        int nt = half * 4 + j;
        acc[0][nt] = wmma_bf16(a0, bf4[j], acc[0][nt]);
        acc[1][nt] = wmma_bf16(a1, bf4[j], acc[1][nt]);
      }
    }
    __syncthreads();
    p ^= 1;
  }

  for (int mt = 0; mt < 2; ++mt)
    for (int nt = 0; nt < 8; ++nt) {
      v8f c = acc[mt][nt];
      for (int v = 0; v < 8; ++v) {
        int m = mbase + mt * 16 + v + 8 * g;   // C layout: M = v + 8*(lane>=16)
        int n = nbase + nt * 16 + ln;          //           N = lane%16
        float val = c[v];
        if (bias) val += bias[n];
        if (mode == M_BF16ROW) {
          ((unsigned short*)out + (long)z * outz)[(long)m * ldo + n] = f2bf(val);
        } else if (mode == M_VT) {             // Vt[b][n][s]
          ((unsigned short*)out)[((long)(m >> 11) * DP + n) * (long)S_ + (m & 2047)] =
              f2bf(val);
        } else if (mode == M_F32SCALE) {
          ((float*)out + (long)z * outz)[(long)m * ldo + n] = val * scale;
        } else {                               // M_GPROJ: g[t][n][b]
          ((float*)out + (long)z * outz)[((long)(m >> 2) * DM + n) * 4 + (m & 3)] = val;
        }
      }
    }
}

// ---------------------------------------------------------------------------
// causal softmax over scores rows -> bf16 attn (zeros above the diagonal)
// one wave per row
// ---------------------------------------------------------------------------
__global__ void softmax_kernel(const float* __restrict__ scores,
                               unsigned short* __restrict__ attn) {
  const int lane = threadIdx.x & 31;
  const long wid = ((long)blockIdx.x * blockDim.x + threadIdx.x) >> 5;
  const int b = (int)(wid >> 11);
  const int s = (int)(wid & 2047);
  const float* row = scores + ((long)b * S_ + s) * S_;
  unsigned short* orow = attn + ((long)b * S_ + s) * S_;
  const int nv = s + 1;
  float mx = -3.0e38f;
  for (int t = lane; t < nv; t += 32) mx = fmaxf(mx, row[t]);
  for (int off = 16; off > 0; off >>= 1) mx = fmaxf(mx, __shfl_xor(mx, off));
  float sum = 0.0f;
  for (int t = lane; t < nv; t += 32) sum += __expf(row[t] - mx);
  for (int off = 16; off > 0; off >>= 1) sum += __shfl_xor(sum, off);
  const float inv = 1.0f / sum;
  for (int t = lane; t < S_; t += 32) {
    float p = (t < nv) ? __expf(row[t] - mx) * inv : 0.0f;
    orow[t] = f2bf(p);
  }
}

// ---------------------------------------------------------------------------
// prefix attention pooling + h0 = tanh(pooled @ Wp2h + b); writes hb rows 0..3
// one block (512 threads) per batch
// ---------------------------------------------------------------------------
__global__ void pool_h0_kernel(const unsigned short* __restrict__ featbf,
                               const float* __restrict__ Wscore,
                               const float* __restrict__ Wp2h,
                               const float* __restrict__ bp2h,
                               unsigned short* __restrict__ hb) {
  __shared__ float sw[AST];
  __shared__ float pooled[DM];
  __shared__ float redv[16];
  const int b = blockIdx.x;
  const int tid = threadIdx.x;
  const unsigned short* fb = featbf + (long)b * S_ * DP;
  const unsigned short* frow = fb + (long)tid * DP;
  float s = 0.0f;
  for (int k = 0; k < DP; ++k) s += bf2f(frow[k]) * Wscore[k];
  float mx = s;
  for (int off = 16; off > 0; off >>= 1) mx = fmaxf(mx, __shfl_xor(mx, off));
  if ((tid & 31) == 0) redv[tid >> 5] = mx;
  __syncthreads();
  float bm = redv[0];
  for (int i = 1; i < 16; ++i) bm = fmaxf(bm, redv[i]);
  float e = __expf(s - bm);
  float se = e;
  for (int off = 16; off > 0; off >>= 1) se += __shfl_xor(se, off);
  __syncthreads();
  if ((tid & 31) == 0) redv[tid >> 5] = se;
  __syncthreads();
  float tot = 0.0f;
  for (int i = 0; i < 16; ++i) tot += redv[i];
  sw[tid] = e / tot;
  __syncthreads();
  float acc = 0.0f;
  for (int u = 0; u < AST; ++u) acc += sw[u] * bf2f(fb[(long)u * DP + tid]);
  pooled[tid] = acc;
  __syncthreads();
  float h = 0.0f;
  for (int k = 0; k < DM; ++k) h += pooled[k] * Wp2h[(long)k * DM + tid];
  h = tanhf(h + bp2h[tid]);
  hb[(long)b * DM + tid] = f2bf(h);   // hb row m = batch index
}

// ---------------------------------------------------------------------------
// GRU scan: 8 persistent WGs, each owns a 64-column slice of Uz/Ur/Uh (bf16,
// [n][k] in LDS, 192 KB, preloaded by the Tensor Data Mover). Batch dim (4)
// padded to M=16. Two grid barriers per step; h and r*h go through L2.
// ---------------------------------------------------------------------------
__device__ inline void gsync(unsigned* ctr, unsigned target, int tid) {
  __syncthreads();
  if (tid == 0) {
    __hip_atomic_fetch_add(ctr, 1u, __ATOMIC_ACQ_REL, __HIP_MEMORY_SCOPE_AGENT);
    while (__hip_atomic_load(ctr, __ATOMIC_ACQUIRE, __HIP_MEMORY_SCOPE_AGENT) < target)
      __builtin_amdgcn_s_sleep(8);
  }
  __syncthreads();
}

__launch_bounds__(128)
__global__ void scan_kernel(const unsigned short* __restrict__ UT3,
                            const float* __restrict__ g3,
                            const float* __restrict__ Wmem,
                            unsigned short* __restrict__ hb,
                            unsigned short* __restrict__ rhb,
                            float* __restrict__ out,
                            unsigned* __restrict__ ctrs) {
  extern __shared__ unsigned short lds[];
  float* red = (float*)(lds + 3 * 64 * DM);
  const int wg   = blockIdx.x;
  const int tid  = threadIdx.x;
  const int w    = tid >> 5;
  const int lane = tid & 31;
  const int g    = lane >> 4;
  const int ln   = lane & 15;
  const int n    = wg * 64 + w * 16 + ln;        // this lane's column

  // TDM preload: one D# moving 3 rows x 64KB (strided 512KB) -> 192KB LDS.
  // data_size=8B: tile_dim0=8192, tile_dim1=3, tensor_dim0_stride=65536.
  if (w == 0) {                                   // wave-uniform branch
    unsigned lds_base = (unsigned)(size_t)(void*)lds;
    unsigned long long ga =
        (unsigned long long)(const void*)(UT3 + (long)wg * 64 * DM);
    u32x4 g0;
    g0.x = 1u;                                          // count=1
    g0.y = lds_base;                                    // lds_addr
    g0.z = (unsigned)(ga & 0xFFFFFFFFull);              // global_addr lo
    g0.w = (unsigned)((ga >> 32) & 0x1FFFFFFull) | (2u << 30);  // hi | type=2
    i32x8 g1;
    g1[0] = 0x00030000;   // data_size=3 (8B units)
    g1[1] = 0x20000000;   // tensor_dim0 = 8192 (bits 63:48)
    g1[2] = 0x00030000;   // tensor_dim1 = 3    (bits 95:80)
    g1[3] = 0x20000000;   // tile_dim0   = 8192 (bits 127:112)
    g1[4] = 0x00000003;   // tile_dim1   = 3    (bits 143:128)
    g1[5] = 0x00010000;   // tensor_dim0_stride = 65536 (bits 191:160)
    g1[6] = 0; g1[7] = 0;
    i32x4 zz4 = {0, 0, 0, 0};
    i32x8 zz8 = {0, 0, 0, 0, 0, 0, 0, 0};
    __builtin_amdgcn_tensor_load_to_lds(g0, g1, zz4, zz4, zz8, 0);
    __builtin_amdgcn_s_wait_tensorcnt(0);
  }
  __syncthreads();

  const float wm_n = Wmem[n];
  const float* gz = g3;
  const float* gr = g3 + (long)RUN * DM * 4;
  const float* gh = g3 + 2L * RUN * DM * 4;
  unsigned* ctrA = ctrs;
  unsigned* ctrB = ctrs + 1;
  const unsigned short* UzS = lds + (long)(w * 16 + ln) * DM;
  const unsigned short* UrS = lds + (long)(64 + w * 16 + ln) * DM;
  const unsigned short* UhS = lds + (long)(128 + w * 16 + ln) * DM;

  for (int t = 0; t < RUN; ++t) {
    // ---- phase 1: z = sig(gz + h@Uz), r = sig(gr + h@Ur), write r*h ----
    v8f za = {}, ra = {};
    for (int kc = 0; kc < DM; kc += 32) {
      FragBF a, bz_, br_;
      const unsigned short* hp = hb + (long)ln * DM + kc;
      a.q[0] = *(const u32x4*)(hp + 8 * g);
      a.q[1] = *(const u32x4*)(hp + 16 + 8 * g);
      const unsigned short* uzp = UzS + kc + 8 * g;
      bz_.q[0] = *(const u32x4*)uzp; bz_.q[1] = *(const u32x4*)(uzp + 16);
      const unsigned short* urp = UrS + kc + 8 * g;
      br_.q[0] = *(const u32x4*)urp; br_.q[1] = *(const u32x4*)(urp + 16);
      za = wmma_bf16(a, bz_, za);
      ra = wmma_bf16(a, br_, ra);
    }
    f32x4 gzv = *(const f32x4*)(gz + ((long)t * DM + n) * 4);
    f32x4 grv = *(const f32x4*)(gr + ((long)t * DM + n) * 4);
    f32x4 ghv = *(const f32x4*)(gh + ((long)t * DM + n) * 4);
    float hprev[4], zq[4];
    for (int v = 0; v < 4; ++v) {
      hprev[v] = (g == 0) ? bf2f(hb[(long)v * DM + n]) : 0.0f;
      float zin = za[v] + ((g == 0) ? gzv[v] : 0.0f);
      float rin = ra[v] + ((g == 0) ? grv[v] : 0.0f);
      zq[v] = 1.0f / (1.0f + __expf(-zin));
      float rv = 1.0f / (1.0f + __expf(-rin));
      if (g == 0) rhb[(long)v * DM + n] = f2bf(rv * hprev[v]);
    }
    __threadfence();
    gsync(ctrA, (unsigned)((t + 1) * NSLICE), tid);

    // ---- phase 2: h_hat = tanh(gh + (r*h)@Uh); update; logit partial ----
    v8f ha = {};
    for (int kc = 0; kc < DM; kc += 32) {
      FragBF a, bh_;
      const unsigned short* rp = rhb + (long)ln * DM + kc;
      a.q[0] = *(const u32x4*)(rp + 8 * g);
      a.q[1] = *(const u32x4*)(rp + 16 + 8 * g);
      const unsigned short* uhp = UhS + kc + 8 * g;
      bh_.q[0] = *(const u32x4*)uhp; bh_.q[1] = *(const u32x4*)(uhp + 16);
      ha = wmma_bf16(a, bh_, ha);
    }
    const float dt = (t == 0) ? 0.0f : (1.0f / 1535.0f);
    float lp[4];
    for (int v = 0; v < 4; ++v) {
      float hh = tanhf(ha[v] + ((g == 0) ? ghv[v] : 0.0f));
      float hn = (1.0f - zq[v]) * hprev[v] + zq[v] * hh;
      float ho = hn + dt * (hn - hprev[v]);
      lp[v] = (g == 0) ? ho * wm_n : 0.0f;
      if (g == 0) hb[(long)v * DM + n] = f2bf(ho);
    }
    for (int off = 16; off > 0; off >>= 1)
      for (int v = 0; v < 4; ++v) lp[v] += __shfl_xor(lp[v], off);
    if (lane == 0) {
      red[w * 4 + 0] = lp[0]; red[w * 4 + 1] = lp[1];
      red[w * 4 + 2] = lp[2]; red[w * 4 + 3] = lp[3];
    }
    __syncthreads();
    if (tid < 4) {
      float s = red[tid] + red[4 + tid] + red[8 + tid] + red[12 + tid];
      atomicAdd(&out[(long)tid * RUN + t], s);
    }
    __threadfence();
    gsync(ctrB, (unsigned)((t + 1) * NSLICE), tid);
  }
}

// ---------------------------------------------------------------------------
// host driver
// ---------------------------------------------------------------------------
extern "C" void kernel_launch(void* const* d_in, const int* in_sizes, int n_in,
                              void* d_out, int out_size, void* d_ws, size_t ws_size,
                              hipStream_t stream) {
  const float* x      = (const float*)d_in[0];
  const float* Wq     = (const float*)d_in[1];
  const float* bq     = (const float*)d_in[2];
  const float* Wk     = (const float*)d_in[3];
  const float* bk     = (const float*)d_in[4];
  const float* Wv     = (const float*)d_in[5];
  const float* bv     = (const float*)d_in[6];
  const float* Wz     = (const float*)d_in[7];
  const float* Uz     = (const float*)d_in[8];
  const float* bz     = (const float*)d_in[9];
  const float* Wr     = (const float*)d_in[10];
  const float* Ur     = (const float*)d_in[11];
  const float* br     = (const float*)d_in[12];
  const float* Wh     = (const float*)d_in[13];
  const float* Uh     = (const float*)d_in[14];
  const float* bh     = (const float*)d_in[15];
  const float* Wmem   = (const float*)d_in[16];
  const float* bmem   = (const float*)d_in[17];
  const float* Wp2h   = (const float*)d_in[18];
  const float* bp2h   = (const float*)d_in[19];
  const float* Wscore = (const float*)d_in[20];
  float* out = (float*)d_out;

  // workspace carve-out (~179 MiB; scores aliases xbf once x is consumed)
  char* ws = (char*)d_ws;
  auto alloc = [&](size_t bytes) -> char* {
    char* p = ws; ws += (bytes + 255) & ~(size_t)255; return p;
  };
  unsigned short* xbf    = (unsigned short*)alloc((size_t)B_ * S_ * DIN * 2);
  unsigned short* WqkvT  = (unsigned short*)alloc((size_t)3 * DP * DIN * 2);
  unsigned short* WgT    = (unsigned short*)alloc((size_t)3 * DM * DP * 2);
  unsigned short* UT3    = (unsigned short*)alloc((size_t)3 * DM * DM * 2);
  float*          bias3  = (float*)alloc((size_t)3 * DP * 4);
  float*          biasg  = (float*)alloc((size_t)3 * DM * 4);
  unsigned short* Qbf    = (unsigned short*)alloc((size_t)B_ * S_ * DP * 2);
  unsigned short* Kbf    = (unsigned short*)alloc((size_t)B_ * S_ * DP * 2);
  unsigned short* Vt     = (unsigned short*)alloc((size_t)B_ * DP * S_ * 2);
  unsigned short* attnbf = (unsigned short*)alloc((size_t)B_ * S_ * S_ * 2);
  unsigned short* featbf = (unsigned short*)alloc((size_t)B_ * S_ * DP * 2);
  float*          g3     = (float*)alloc((size_t)3 * RUN * DM * 4 * 4);
  unsigned short* hb     = (unsigned short*)alloc((size_t)16 * DM * 2);
  unsigned short* rhb    = (unsigned short*)alloc((size_t)16 * DM * 2);
  unsigned*       ctrs   = (unsigned*)alloc(256);
  float* scoresf = (float*)xbf;   // alias: xbf dead after QKV GEMMs

  // 0) init output(+bmem), h/rh buffers, sync counters
  init_kernel<<<32, 256, 0, stream>>>(out, bmem, hb, rhb, ctrs);

  // 1) fp32 -> bf16 conversions / transposes
  conv_bf16_kernel<<<(unsigned)((long)B_ * S_ * DIN / 4 / 256), 256, 0, stream>>>(
      x, xbf, (long)B_ * S_ * DIN / 4);
  transpose_bf16_kernel<<<dim3(DIN / 32, DP / 32), 256, 0, stream>>>(Wq, WqkvT, DIN, DP);
  transpose_bf16_kernel<<<dim3(DIN / 32, DP / 32), 256, 0, stream>>>(Wk, WqkvT + (size_t)DP * DIN, DIN, DP);
  transpose_bf16_kernel<<<dim3(DIN / 32, DP / 32), 256, 0, stream>>>(Wv, WqkvT + (size_t)2 * DP * DIN, DIN, DP);
  transpose_bf16_kernel<<<dim3(DP / 32, DM / 32), 256, 0, stream>>>(Wz, WgT, DP, DM);
  transpose_bf16_kernel<<<dim3(DP / 32, DM / 32), 256, 0, stream>>>(Wr, WgT + (size_t)DM * DP, DP, DM);
  transpose_bf16_kernel<<<dim3(DP / 32, DM / 32), 256, 0, stream>>>(Wh, WgT + (size_t)2 * DM * DP, DP, DM);
  transpose_bf16_kernel<<<dim3(DM / 32, DM / 32), 256, 0, stream>>>(Uz, UT3, DM, DM);
  transpose_bf16_kernel<<<dim3(DM / 32, DM / 32), 256, 0, stream>>>(Ur, UT3 + (size_t)DM * DM, DM, DM);
  transpose_bf16_kernel<<<dim3(DM / 32, DM / 32), 256, 0, stream>>>(Uh, UT3 + (size_t)2 * DM * DM, DM, DM);
  bias_stage_kernel<<<2, 256, 0, stream>>>(bq, bk, bv, bz, br, bh, bias3, biasg);

  // 2) Q/K/V projections (M=8192, N=512, K=4096)
  dim3 gq(B_ * S_ / 256, DP / 128, 1);
  gemm_bf16_kernel<<<gq, 256, 0, stream>>>(xbf, DIN, 0, WqkvT, DIN, 0,
      bias3, 0, Qbf, 0, DP, DIN, M_BF16ROW, 1.0f);
  gemm_bf16_kernel<<<gq, 256, 0, stream>>>(xbf, DIN, 0, WqkvT + (size_t)DP * DIN, DIN, 0,
      bias3 + DP, 0, Kbf, 0, DP, DIN, M_BF16ROW, 1.0f);
  gemm_bf16_kernel<<<gq, 256, 0, stream>>>(xbf, DIN, 0, WqkvT + (size_t)2 * DP * DIN, DIN, 0,
      bias3 + 2 * DP, 0, Vt, 0, DP, DIN, M_VT, 1.0f);

  // 3) scores = Q K^T / sqrt(Dp)  (per batch, M=N=2048, K=512)
  const float scale = 1.0f / sqrtf((float)DP + 1e-6f);
  dim3 gs(S_ / 256, S_ / 128, B_);
  gemm_bf16_kernel<<<gs, 256, 0, stream>>>(Qbf, DP, (long)S_ * DP, Kbf, DP, (long)S_ * DP,
      nullptr, 0, scoresf, (long)S_ * S_, S_, DP, M_F32SCALE, scale);

  // 4) causal softmax -> bf16 attention
  softmax_kernel<<<(B_ * S_ * 32) / 256, 256, 0, stream>>>(scoresf, attnbf);

  // 5) feat = attn @ V  (per batch, M=2048, N=512, K=2048)
  dim3 gf(S_ / 256, DP / 128, B_);
  gemm_bf16_kernel<<<gf, 256, 0, stream>>>(attnbf, S_, (long)S_ * S_, Vt, S_, (long)DP * S_,
      nullptr, 0, featbf, (long)S_ * DP, DP, S_, M_BF16ROW, 1.0f);

  // 6) prefix pooling -> h0 (writes hb rows 0..3)
  pool_h0_kernel<<<B_, 512, 0, stream>>>(featbf, Wscore, Wp2h, bp2h, hb);

  // 7) GRU input projections: g[mat][t][n][b] = xs @ W + b  (M=6144,N=512,K=512)
  dim3 gg(B_ * RUN / 256, DM / 128, 3);
  gemm_bf16_kernel<<<gg, 256, 0, stream>>>(featbf, DP, 0, WgT, DP, (long)DM * DP,
      biasg, DM, g3, (long)RUN * DM * 4, 0, DP, M_GPROJ, 1.0f);

  // 8) persistent cooperative GRU scan (8 WGs, 192 KB LDS each, TDM preload)
  const int lds_bytes = 3 * 64 * DM * 2 + 256;
  (void)hipFuncSetAttribute(reinterpret_cast<const void*>(scan_kernel),
                            hipFuncAttributeMaxDynamicSharedMemorySize, lds_bytes);
  scan_kernel<<<NSLICE, 128, lds_bytes, stream>>>(UT3, g3, Wmem, hb, rhb, out, ctrs);

  (void)in_sizes; (void)n_in; (void)out_size; (void)ws_size;
}
